// GATConv_69080253988970
// MI455X (gfx1250) — matrix operand order repaired and checked by reading, not compiled, not run on hardware
//
#include <hip/hip_runtime.h>
#include <hip/hip_bf16.h>

// ---------------- problem constants (match reference) ----------------
#define N_NODES 50000
#define E_EDGES 1600000
#define F_IN 64
#define HC 64            // H*C
#define HEADS 4
#define NEG_SLOPE 0.2f

#define N_TILES (N_NODES / 16)   // 3125
#define E_TILES (E_EDGES / 16)   // 100000

// Wp holds 2 K-steps x 5 B-tiles x 32 lanes x 16 bf16 = 5120 elems (10 KB).
// Tiles 0..3: columns of W (head blocks). Tile 4: projection matrix P:
//   P[:,0..3]  = W . dst_attn  (per head)   -> edge logit term
//   P[:,4..7]  = W . src_attn               -> al_sum (gathered logit term)
//   P[:,8..11] = W . (src_attn + dst_attn)  -> self logit
//   P[:,12..15]= 0
#define WP_ELEMS (2 * 5 * 32 * 16)

typedef __attribute__((ext_vector_type(16))) __bf16 v16bf;
typedef __attribute__((ext_vector_type(8)))  float  v8f;
typedef __attribute__((ext_vector_type(4)))  float  v4f;

static __device__ __forceinline__ __bf16 f2bf(float f) { return (__bf16)f; }

template <bool NT>
static __device__ __forceinline__ v4f ld4(const float* p) {
    if (NT) return __builtin_nontemporal_load((const v4f*)p);
    return *(const v4f*)p;
}

// ds_swizzle BCASTX16: src_lane = (lane & 0x10) | SRC
// offset encoding: [14:10]=xor, [9:5]=or, [4:0]=and  (immediate must be constexpr)
template <int SRC>
static __device__ __forceinline__ float bcast16(float v) {
    return __int_as_float(
        __builtin_amdgcn_ds_swizzle(__float_as_int(v), 0x10 | (SRC << 5)));
}

// -------------------------------------------------------------------
// Prep kernel: build bf16 WMMA B-operand register images for W and P.
// Slot t = ((ks*5 + nt)*32 + lane); element j -> K = ks*32 + 16*(lane>=16) + j,
// column = nt*16 + (lane&15) for W tiles, column = (lane&15) for P tile.
// -------------------------------------------------------------------
__global__ void __launch_bounds__(320) gat_prep_W(
    const float* __restrict__ W, const float* __restrict__ src_attn,
    const float* __restrict__ dst_attn, __bf16* __restrict__ Wp)
{
    const int t = threadIdx.x;          // 320 threads, 1 block
    const int ks   = t / 160;
    const int nt   = (t / 32) % 5;
    const int lane = t & 31;
    const int lc   = lane & 15;
    const int kb   = ks * 32 + (lane >> 4) * 16;
    __bf16* o = Wp + (long)t * 16;

    if (nt < 4) {
        const int col = nt * 16 + lc;
        #pragma unroll
        for (int j = 0; j < 16; ++j)
            o[j] = f2bf(W[(kb + j) * HC + col]);
    } else {
        for (int j = 0; j < 16; ++j) {
            float v = 0.0f;
            if (lc < 12) {
                const int h = lc & 3;
                const int K = kb + j;
                #pragma unroll
                for (int c = 0; c < 16; ++c) {
                    const float wv = W[K * HC + h * 16 + c];
                    float a;
                    if (lc < 4)      a = dst_attn[h * 16 + c];
                    else if (lc < 8) a = src_attn[h * 16 + c];
                    else             a = src_attn[h * 16 + c] + dst_attn[h * 16 + c];
                    v += wv * a;
                }
            }
            o[j] = f2bf(v);
        }
    }
}

// Load pre-swizzled B operands: 10 aligned 32-byte vector loads.
static __device__ __forceinline__ void load_B_tiles(const __bf16* __restrict__ Wp,
                                                    int lane, v16bf Bm[2][5]) {
    #pragma unroll
    for (int ks = 0; ks < 2; ++ks)
        #pragma unroll
        for (int nt = 0; nt < 5; ++nt)
            Bm[ks][nt] = *(const v16bf*)(Wp + (long)((ks * 5 + nt) * 32 + lane) * 16);
}

// Load A-operand (16 rows x 64 cols f32, row major -> bf16 WMMA layout).
template <bool NT>
static __device__ __forceinline__ void load_A_tiles(const float* __restrict__ rows,
                                                    long rowBase, int lane, v16bf Am[2]) {
    const int half = lane >> 4;
    const int lc   = lane & 15;
    const float* rp = rows + (rowBase + lc) * F_IN;
    #pragma unroll
    for (int ks = 0; ks < 2; ++ks) {
        const int kb = ks * 32 + half * 16;
        const v4f c0 = ld4<NT>(rp + kb);
        const v4f c1 = ld4<NT>(rp + kb + 4);
        const v4f c2 = ld4<NT>(rp + kb + 8);
        const v4f c3 = ld4<NT>(rp + kb + 12);
        v16bf a;
        #pragma unroll
        for (int j = 0; j < 4; ++j) {
            a[j]      = f2bf(c0[j]);
            a[4 + j]  = f2bf(c1[j]);
            a[8 + j]  = f2bf(c2[j]);
            a[12 + j] = f2bf(c3[j]);
        }
        Am[ks] = a;
    }
}

// -------------------------------------------------------------------
// Kernel 0: zero the accumulators (acc [N,64] and norm [N,4], contiguous)
// -------------------------------------------------------------------
__global__ void __launch_bounds__(256) gat_zero_kernel(float* __restrict__ p, long n) {
    long i = (long)blockIdx.x * blockDim.x + threadIdx.x;
    const long stride = (long)gridDim.x * blockDim.x;
    for (; i < n; i += stride) p[i] = 0.0f;
}

// -------------------------------------------------------------------
// Kernel 1: node side.  D[0..3] = x_src tile, E = [logit projections].
//   al_sum[n,h] = E col 4+h ;  self_w[n,h] = exp(leaky(E col 8+h))
// -------------------------------------------------------------------
__global__ void __launch_bounds__(256) gat_node_kernel(
    const float* __restrict__ src, const __bf16* __restrict__ Wp,
    float* __restrict__ x_src, float* __restrict__ al_sum, float* __restrict__ self_w)
{
    const int lane = threadIdx.x & 31;
    const int tile = blockIdx.x * 8 + (threadIdx.x >> 5);
    if (tile >= N_TILES) return;                    // wave-uniform
    const int half = lane >> 4;
    const int lc   = lane & 15;
    const long rowBase = (long)tile * 16;

    v16bf Bm[2][5], Am[2];
    load_B_tiles(Wp, lane, Bm);
    load_A_tiles<false>(src, rowBase, lane, Am);

    v8f D[4], E;
    #pragma unroll
    for (int nt = 0; nt < 4; ++nt) {
        v8f c = {};
        c = __builtin_amdgcn_wmma_f32_16x16x32_bf16(false, Am[0], false, Bm[0][nt],
                                                    (short)0, c, false, false);
        c = __builtin_amdgcn_wmma_f32_16x16x32_bf16(false, Am[1], false, Bm[1][nt],
                                                    (short)0, c, false, false);
        D[nt] = c;
    }
    {
        v8f c = {};
        c = __builtin_amdgcn_wmma_f32_16x16x32_bf16(false, Am[0], false, Bm[0][4],
                                                    (short)0, c, false, false);
        c = __builtin_amdgcn_wmma_f32_16x16x32_bf16(false, Am[1], false, Bm[1][4],
                                                    (short)0, c, false, false);
        E = c;
    }

    // store x_src: element r at lane l = row r+8*half, col nt*16+lc
    #pragma unroll
    for (int nt = 0; nt < 4; ++nt) {
        #pragma unroll
        for (int r = 0; r < 8; ++r)
            x_src[(rowBase + r + 8 * half) * HC + nt * 16 + lc] = D[nt][r];
    }

    // E columns carry the reduced logit terms directly
    if (lc >= 4 && lc < 8) {
        const int h = lc - 4;
        #pragma unroll
        for (int r = 0; r < 8; ++r)
            al_sum[(rowBase + r + 8 * half) * HEADS + h] = E[r];
    } else if (lc >= 8 && lc < 12) {
        const int h = lc - 8;
        #pragma unroll
        for (int r = 0; r < 8; ++r) {
            const float lg = E[r];
            const float lr = lg > 0.0f ? lg : NEG_SLOPE * lg;
            self_w[(rowBase + r + 8 * half) * HEADS + h] = __expf(lr);
        }
    }
}

// Per-head scatter for the edge kernel (HEAD is a compile-time constant so the
// ds_swizzle immediate is a constexpr).
template <int HEAD>
static __device__ __forceinline__ void edge_scatter_head(
    const float w[8], const v8f& Dh, const int segs[8], int lc,
    float* __restrict__ acc, float* __restrict__ norm)
{
    // broadcast w[row][HEAD] from lane (half*16 + HEAD) to all lanes of the half
    float wb[8];
    #pragma unroll
    for (int r = 0; r < 8; ++r) wb[r] = bcast16<HEAD>(w[r]);

    // feature scatter: combine runs of equal segment before atomics
    {
        float pend = 0.0f; int pseg = segs[0];
        #pragma unroll
        for (int r = 0; r < 8; ++r) {
            if (segs[r] != pseg) {
                atomicAdd(&acc[(long)pseg * HC + HEAD * 16 + lc], pend);
                pend = 0.0f; pseg = segs[r];
            }
            pend += wb[r] * Dh[r];
        }
        atomicAdd(&acc[(long)pseg * HC + HEAD * 16 + lc], pend);
    }
    // normalizer scatter (one lane per half)
    if (lc == 0) {
        float pend = 0.0f; int pseg = segs[0];
        #pragma unroll
        for (int r = 0; r < 8; ++r) {
            if (segs[r] != pseg) {
                atomicAdd(&norm[(long)pseg * HEADS + HEAD], pend);
                pend = 0.0f; pseg = segs[r];
            }
            pend += wb[r];
        }
        atomicAdd(&norm[(long)pseg * HEADS + HEAD], pend);
    }
}

// -------------------------------------------------------------------
// Kernel 2: edge side.  D[0..3] = x_dst tile, E cols 0..3 = dst-attn term.
//   w = exp(leaky(E[.,h] + al_sum[seg,h])); run-combined atomics:
//   norm[seg,h] += w ;  acc[seg,h*16+c] += w * x_dst[h,c]
// -------------------------------------------------------------------
__global__ void __launch_bounds__(256) gat_edge_kernel(
    const float* __restrict__ dst, const int* __restrict__ seg,
    const __bf16* __restrict__ Wp, const float* __restrict__ al_sum,
    float* __restrict__ acc, float* __restrict__ norm)
{
    const int lane = threadIdx.x & 31;
    const int tile = blockIdx.x * 8 + (threadIdx.x >> 5);
    if (tile >= E_TILES) return;                    // wave-uniform
    const int half = lane >> 4;
    const int lc   = lane & 15;
    const long edgeBase = (long)tile * 16;

    v16bf Bm[2][5], Am[2];
    load_B_tiles(Wp, lane, Bm);
    load_A_tiles<true>(dst, edgeBase, lane, Am);    // non-temporal: dst streamed once

    v8f D[4], E;
    #pragma unroll
    for (int nt = 0; nt < 4; ++nt) {
        v8f c = {};
        c = __builtin_amdgcn_wmma_f32_16x16x32_bf16(false, Am[0], false, Bm[0][nt],
                                                    (short)0, c, false, false);
        c = __builtin_amdgcn_wmma_f32_16x16x32_bf16(false, Am[1], false, Bm[1][nt],
                                                    (short)0, c, false, false);
        D[nt] = c;
    }
    {
        v8f c = {};
        c = __builtin_amdgcn_wmma_f32_16x16x32_bf16(false, Am[0], false, Bm[0][4],
                                                    (short)0, c, false, false);
        c = __builtin_amdgcn_wmma_f32_16x16x32_bf16(false, Am[1], false, Bm[1][4],
                                                    (short)0, c, false, false);
        E = c;
    }

    // segment ids for this lane's 8 rows (rows half*8 + r); sorted -> half-uniform
    int segs[8];
    #pragma unroll
    for (int r = 0; r < 8; ++r) segs[r] = seg[edgeBase + half * 8 + r];

    // per-edge unnormalized weight for head (lc & 3); valid on lanes lc<4,
    // harmless elsewhere (E cols 4..11 are finite projections, 12..15 zero).
    const int hme = lc & 3;
    float w[8];
    #pragma unroll
    for (int r = 0; r < 8; ++r) {
        const float lg = E[r] + al_sum[(long)segs[r] * HEADS + hme];
        const float lr = lg > 0.0f ? lg : NEG_SLOPE * lg;
        w[r] = __expf(lr);
    }

    edge_scatter_head<0>(w, D[0], segs, lc, acc, norm);
    edge_scatter_head<1>(w, D[1], segs, lc, acc, norm);
    edge_scatter_head<2>(w, D[2], segs, lc, acc, norm);
    edge_scatter_head<3>(w, D[3], segs, lc, acc, norm);
}

// -------------------------------------------------------------------
// Kernel 3: finalize.  out = (acc + self_w * x_src) / (norm + self_w)
// -------------------------------------------------------------------
__global__ void __launch_bounds__(256) gat_finalize_kernel(
    const float* __restrict__ acc, const float* __restrict__ norm,
    const float* __restrict__ x_src, const float* __restrict__ self_w,
    float* __restrict__ out)
{
    const long i = (long)blockIdx.x * blockDim.x + threadIdx.x;
    if (i >= (long)N_NODES * HC) return;
    const long n = i >> 6;
    const int  h = (int)((i >> 4) & 3);
    const float sw  = self_w[n * HEADS + h];
    const float den = norm[n * HEADS + h] + sw;
    out[i] = (acc[i] + sw * x_src[i]) / den;
}

// -------------------------------------------------------------------
extern "C" void kernel_launch(void* const* d_in, const int* in_sizes, int n_in,
                              void* d_out, int out_size, void* d_ws, size_t ws_size,
                              hipStream_t stream) {
    const float* src      = (const float*)d_in[0];   // [N,64]
    const float* dst      = (const float*)d_in[1];   // [E,64]
    const int*   seg      = (const int*)  d_in[2];   // [E]
    const float* W        = (const float*)d_in[3];   // [64,64]
    const float* src_attn = (const float*)d_in[4];   // [1,4,16]
    const float* dst_attn = (const float*)d_in[5];   // [1,4,16]
    float* out = (float*)d_out;                      // [N,64]

    // workspace layout: Wp (bf16, 5120 elems = 10KB) | x_src | acc | norm | al_sum | self_w
    __bf16* Wp    = (__bf16*)d_ws;
    float* ws_f   = (float*)((char*)d_ws + WP_ELEMS * sizeof(__bf16));
    float* x_src  = ws_f;                                // N*64
    float* acc    = x_src + (long)N_NODES * HC;          // N*64
    float* norm   = acc   + (long)N_NODES * HC;          // N*4
    float* al_sum = norm  + (long)N_NODES * HEADS;       // N*4
    float* self_w = al_sum+ (long)N_NODES * HEADS;       // N*4

    // 0) swizzle W (+ attention projections) into bf16 WMMA layout
    gat_prep_W<<<1, 320, 0, stream>>>(W, src_attn, dst_attn, Wp);

    // 1) zero acc+norm (contiguous)
    const long zn = (long)N_NODES * (HC + HEADS);
    gat_zero_kernel<<<2048, 256, 0, stream>>>(acc, zn);

    // 2) node projection + self terms (WMMA)
    gat_node_kernel<<<(N_TILES + 7) / 8, 256, 0, stream>>>(
        src, Wp, x_src, al_sum, self_w);

    // 3) edge projection + attention + scatter-aggregate (WMMA)
    gat_edge_kernel<<<(E_TILES + 7) / 8, 256, 0, stream>>>(
        dst, seg, Wp, al_sum, acc, norm);

    // 4) normalize + self term
    const long total = (long)N_NODES * HC;
    gat_finalize_kernel<<<(int)((total + 255) / 256), 256, 0, stream>>>(
        acc, norm, x_src, self_w, out);
}
